// MOE_56925496541300
// MI455X (gfx1250) — compile-verified, matrix operand order checked by reading, not run on hardware
//
#include <hip/hip_runtime.h>
#include <hip/hip_bf16.h>
#include <math.h>

// ---------------------------------------------------------------------------
// MoE transformer for MI455X (gfx1250, wave32, WMMA).
// Heavy GEMMs: v_wmma_f32_16x16x32_bf16, register-blocked 32x64 per wave,
// B tiles staged block-wide in LDS via global_load_async_to_lds_b128
// (ASYNCcnt-tracked async DMA), double buffered.
// ---------------------------------------------------------------------------

typedef __attribute__((ext_vector_type(16))) __bf16 v16bf;
typedef __attribute__((ext_vector_type(8)))  __bf16 v8bf;
typedef __attribute__((ext_vector_type(8)))  float  v8f;

// Model constants (match reference)
#define TB    8      // batch
#define TS    1024   // seq
#define TD    256    // d_model
#define TH    8      // heads
#define THD   32     // head dim
#define TFF   1024   // expert hidden
#define TE    8      // experts
#define TL    2      // layers
#define TOUT  1000
#define TTOK  (TB * TS)   // 8192 tokens

// ---------------------------------------------------------------------------
// Fragment load for 16x32 bf16 A/B tile (BT stored [N x K] row-major so B uses
// the identical loader; also used on LDS tiles). ISA 16-bit A layout:
// lane = row (mod 16), lane-half selects K offset +0/+8, element-group selects
// K offset +0/+16.  Each lane = two contiguous 16-byte loads.
// ---------------------------------------------------------------------------
__device__ inline v16bf load_frag(const __bf16* base, int ld, int row0, int k0) {
  const int lane = threadIdx.x & 31;
  const int r    = row0 + (lane & 15);
  const int half = (lane >> 4) & 1;
  const __bf16* p0 = base + (size_t)r * ld + k0 + half * 8;
  v8bf lo = *(const v8bf*)(p0);
  v8bf hi = *(const v8bf*)(p0 + 16);
  v16bf f;
#pragma unroll
  for (int i = 0; i < 8; ++i) { f[i] = lo[i]; f[i + 8] = hi[i]; }
  return f;
}

__device__ inline v8f wmma_bf16(v16bf a, v16bf b, v8f c) {
  return __builtin_amdgcn_wmma_f32_16x16x32_bf16(false, a, false, b, (short)0, c,
                                                 false, false);
}

// Async global->LDS copy: each lane moves 16B; LDS byte address in VDST VGPR,
// 64-bit global address in VADDR pair. Tracked by ASYNCcnt.
__device__ inline void async_b128(unsigned laddr, unsigned long long gaddr) {
  asm volatile("global_load_async_to_lds_b128 %0, %1, off"
               :: "v"(laddr), "v"(gaddr)
               : "memory");
}
__device__ inline void wait_async0() {
  asm volatile("s_wait_asynccnt 0x0" ::: "memory");
}

// ---------------------------------------------------------------------------
// Register-blocked WMMA GEMM: C[M,N] = A[M,K](bf16) * BT[N,K]^T(bf16) + bias.
// Each wave owns a 32x64 macro-tile: 2 (M) x 4 (N) 16x16 accumulators; per
// 32-wide K-step: 4 global b128 A-loads + 8 LDS b128 B-loads -> 8 WMMAs.
// The 64x32 B tile (4KB) is copied to LDS once per block with 8 async-DMA
// instructions (2 per wave), double-buffered against compute.
// NOTE: all launches have M%128==0 and N%64==0, so EXEC is all-ones at every
// WMMA and no wave skips the barriers.
// Fused epilogues:
//   EPI 0: out = bf16(acc + bias)                       (QKV)
//   EPI 1: xout += acc + bias                           (attn out-proj residual)
//   EPI 2: out = bf16(gelu_exact(acc + bias))           (expert w1)
//   EPI 3: xout += coef[row] * (acc + bias)             (expert w2 + MoE combine)
// ---------------------------------------------------------------------------
template <int EPI>
__global__ __launch_bounds__(128) void gemm_wmma(
    const __bf16* __restrict__ A, int lda,
    const __bf16* __restrict__ BT, int ldbt,
    const float* __restrict__ bias,
    float* __restrict__ xout, __bf16* __restrict__ obf, int ldo,
    const float* __restrict__ coef,
    int M, int N, int K)
{
  __shared__ __align__(16) __bf16 Bls[2][64 * 32];   // 2 x 4KB
  const int lane = threadIdx.x;
  const int tid  = threadIdx.y * 32 + lane;          // 0..127
  const int m0 = blockIdx.y * 128 + threadIdx.y * 32;  // rows m0 .. m0+31
  const int n0 = blockIdx.x * 64;                      // cols n0 .. n0+63

  // This thread's two 16B chunks of the 4KB B tile (row stride 64B = 32 bf16).
  const int o0 = tid * 16;          // byte offset in tile
  const int o1 = o0 + 2048;
  const int r0 = o0 >> 6, c0 = (o0 & 63) >> 1;
  const int r1 = o1 >> 6, c1 = (o1 & 63) >> 1;
  const unsigned lbase = (unsigned)(size_t)(&Bls[0][0]);

  // prologue: stage first B tile
  {
    const __bf16* g0 = BT + (size_t)(n0 + r0) * ldbt + 0 + c0;
    const __bf16* g1 = BT + (size_t)(n0 + r1) * ldbt + 0 + c1;
    async_b128(lbase + o0, (unsigned long long)g0);
    async_b128(lbase + o1, (unsigned long long)g1);
  }

  v8f acc[2][4];
#pragma unroll
  for (int i = 0; i < 2; ++i)
#pragma unroll
    for (int j = 0; j < 4; ++j) acc[i][j] = (v8f){};

  int buf = 0;
  for (int k0 = 0; k0 < K; k0 += 32) {
    wait_async0();        // this wave's copies into Bls[buf] have landed
    __syncthreads();      // everyone's copies landed; prior reads of Bls[buf^1] done
    if (k0 + 32 < K) {    // prefetch next tile into the other buffer
      const __bf16* g0 = BT + (size_t)(n0 + r0) * ldbt + (k0 + 32) + c0;
      const __bf16* g1 = BT + (size_t)(n0 + r1) * ldbt + (k0 + 32) + c1;
      unsigned lb = lbase + (unsigned)((buf ^ 1) * 4096);
      async_b128(lb + o0, (unsigned long long)g0);
      async_b128(lb + o1, (unsigned long long)g1);
    }
    v16bf a0 = load_frag(A, lda, m0, k0);
    v16bf a1 = load_frag(A, lda, m0 + 16, k0);
    const __bf16* Bc = &Bls[buf][0];
    v16bf b0 = load_frag(Bc, 32, 0, 0);
    v16bf b1 = load_frag(Bc, 32, 16, 0);
    v16bf b2 = load_frag(Bc, 32, 32, 0);
    v16bf b3 = load_frag(Bc, 32, 48, 0);
    acc[0][0] = wmma_bf16(a0, b0, acc[0][0]);
    acc[0][1] = wmma_bf16(a0, b1, acc[0][1]);
    acc[0][2] = wmma_bf16(a0, b2, acc[0][2]);
    acc[0][3] = wmma_bf16(a0, b3, acc[0][3]);
    acc[1][0] = wmma_bf16(a1, b0, acc[1][0]);
    acc[1][1] = wmma_bf16(a1, b1, acc[1][1]);
    acc[1][2] = wmma_bf16(a1, b2, acc[1][2]);
    acc[1][3] = wmma_bf16(a1, b3, acc[1][3]);
    buf ^= 1;
  }

  const int cl = lane & 15;
  const int rh = 8 * (lane >> 4);
#pragma unroll
  for (int j = 0; j < 4; ++j) {
    const int col = n0 + j * 16 + cl;
    const float bc = bias[col];
#pragma unroll
    for (int i = 0; i < 2; ++i) {
#pragma unroll
      for (int v = 0; v < 8; ++v) {
        const int row = m0 + i * 16 + rh + v;
        float r = acc[i][j][v] + bc;
        if (EPI == 0) {
          obf[(size_t)row * ldo + col] = (__bf16)r;
        } else if (EPI == 1) {
          xout[(size_t)row * ldo + col] += r;
        } else if (EPI == 2) {
          float g = 0.5f * r * (1.0f + erff(r * 0.70710678118654752f));
          obf[(size_t)row * ldo + col] = (__bf16)g;
        } else {
          float c = coef[(size_t)row * TE];   // coef pre-offset by expert index
          xout[(size_t)row * ldo + col] += c * r;
        }
      }
    }
  }
}

// ---------------------------------------------------------------------------
// Flash attention. One wave per (b, h, 16-query tile). HD=32 == WMMA K, so a
// 16x16 score tile is one WMMA. Online softmax with row stats in C-fragment
// layout (row = v + 8*(lane>=16)); reductions via shfl_xor over 16-lane halves.
// P is converted fp32->bf16 through LDS to reshape C-frag -> A-frag for P*V.
// ---------------------------------------------------------------------------
__global__ __launch_bounds__(32) void attn_kernel(const __bf16* __restrict__ qkv,
                                                  __bf16* __restrict__ attn_o)
{
  __shared__ __align__(16) __bf16 Pls[16 * 32];
  const int lane = threadIdx.x;
  const int b    = blockIdx.x >> 3;
  const int h    = blockIdx.x & 7;
  const int q0   = blockIdx.y * 16;
  const int ld   = 3 * TD;            // 768
  const int brow = b * TS;
  const float scale = 0.17677669529663687f;   // 1/sqrt(32)

  v16bf qf = load_frag(qkv + h * THD, ld, brow + q0, 0);

  v8f o0 = {}, o1 = {};
  float mrow[8], lrow[8];
#pragma unroll
  for (int v = 0; v < 8; ++v) { mrow[v] = -1e30f; lrow[v] = 0.0f; }

  for (int kb = 0; kb < TS; kb += 32) {
    // scores: Q(16x32) x K^T -> two 16x16 tiles (K stored row-major == BT form)
    v16bf kf0 = load_frag(qkv + TD + h * THD, ld, brow + kb, 0);
    v16bf kf1 = load_frag(qkv + TD + h * THD, ld, brow + kb + 16, 0);
    v8f s0 = {}, s1 = {};
    s0 = wmma_bf16(qf, kf0, s0);
    s1 = wmma_bf16(qf, kf1, s1);

    float p0[8], p1[8];
#pragma unroll
    for (int v = 0; v < 8; ++v) {
      float a  = s0[v] * scale;
      float bb = s1[v] * scale;
      float mx = fmaxf(a, bb);
#pragma unroll
      for (int msk = 1; msk <= 8; msk <<= 1) mx = fmaxf(mx, __shfl_xor(mx, msk, 32));
      float mnew  = fmaxf(mrow[v], mx);
      float alpha = __expf(mrow[v] - mnew);
      a  = __expf(a - mnew);
      bb = __expf(bb - mnew);
      float rs = a + bb;
#pragma unroll
      for (int msk = 1; msk <= 8; msk <<= 1) rs += __shfl_xor(rs, msk, 32);
      lrow[v] = lrow[v] * alpha + rs;
      mrow[v] = mnew;
      o0[v] *= alpha;
      o1[v] *= alpha;
      p0[v] = a;
      p1[v] = bb;
    }

    // C-frag -> LDS (bf16) -> A-frag reshape for P
    {
      const int colp  = lane & 15;
      const int rbase = 8 * (lane >> 4);
#pragma unroll
      for (int v = 0; v < 8; ++v) {
        Pls[(rbase + v) * 32 + colp]      = (__bf16)p0[v];
        Pls[(rbase + v) * 32 + colp + 16] = (__bf16)p1[v];
      }
    }
    __syncthreads();
    v16bf pf = load_frag(Pls, 32, 0, 0);

    // V fragments: need V^T layout (lane = hd col, elements over key) -> gather
    v16bf vf0, vf1;
    {
      const int n    = lane & 15;
      const int half = lane >> 4;
#pragma unroll
      for (int e = 0; e < 16; ++e) {
        int key = kb + (e >> 3) * 16 + half * 8 + (e & 7);
        size_t base = (size_t)(brow + key) * ld + 2 * TD + h * THD;
        vf0[e] = qkv[base + n];
        vf1[e] = qkv[base + 16 + n];
      }
    }
    o0 = wmma_bf16(pf, vf0, o0);
    o1 = wmma_bf16(pf, vf1, o1);
    __syncthreads();
  }

  const int colp  = lane & 15;
  const int rbase = 8 * (lane >> 4);
#pragma unroll
  for (int v = 0; v < 8; ++v) {
    float inv = 1.0f / lrow[v];
    size_t row = (size_t)(brow + q0 + rbase + v);
    attn_o[row * TD + h * THD + colp]      = (__bf16)(o0[v] * inv);
    attn_o[row * TD + h * THD + 16 + colp] = (__bf16)(o1[v] * inv);
  }
}

// ---------------------------------------------------------------------------
// LayerNorm: one 256-thread block per token, fp32 in -> bf16 out.
// ---------------------------------------------------------------------------
__global__ __launch_bounds__(256) void ln_kernel(const float* __restrict__ x,
                                                 const float* __restrict__ g,
                                                 const float* __restrict__ be,
                                                 __bf16* __restrict__ out)
{
  __shared__ float s1[8], s2[8];
  const int t = blockIdx.x;
  const int d = threadIdx.x;
  float v = x[(size_t)t * TD + d];
  float a = v, b2 = v * v;
#pragma unroll
  for (int m = 16; m >= 1; m >>= 1) {
    a  += __shfl_xor(a, m, 32);
    b2 += __shfl_xor(b2, m, 32);
  }
  const int wid = d >> 5;
  if ((d & 31) == 0) { s1[wid] = a; s2[wid] = b2; }
  __syncthreads();
  float tot = 0.f, tot2 = 0.f;
#pragma unroll
  for (int i = 0; i < 8; ++i) { tot += s1[i]; tot2 += s2[i]; }
  float mean = tot * (1.0f / TD);
  float var  = tot2 * (1.0f / TD) - mean * mean;
  float rstd = rsqrtf(var + 1e-5f);
  out[(size_t)t * TD + d] = (__bf16)((v - mean) * rstd * g[d] + be[d]);
}

// ---------------------------------------------------------------------------
// Gating: per token, 8 gate logits -> top-2 -> softmax; writes coef[T,8]
// (zero for unselected experts).
// ---------------------------------------------------------------------------
__global__ void gate_kernel(const __bf16* __restrict__ xln,
                            const float* __restrict__ gw,
                            const float* __restrict__ gb,
                            float* __restrict__ coef, int T)
{
  int t = blockIdx.x * blockDim.x + threadIdx.x;
  if (t >= T) return;
  float g[TE];
#pragma unroll
  for (int e = 0; e < TE; ++e) g[e] = gb[e];
  for (int d = 0; d < TD; ++d) {
    float xv = (float)xln[(size_t)t * TD + d];
    const float* w = gw + d * TE;
#pragma unroll
    for (int e = 0; e < TE; ++e) g[e] += xv * w[e];
  }
  int i0 = 0;
#pragma unroll
  for (int e = 1; e < TE; ++e) if (g[e] > g[i0]) i0 = e;
  int i1 = (i0 == 0) ? 1 : 0;
#pragma unroll
  for (int e = 0; e < TE; ++e) if (e != i0 && g[e] > g[i1]) i1 = e;
  float e1 = __expf(g[i1] - g[i0]);
  float w0 = 1.0f / (1.0f + e1);
  float w1 = e1 * w0;
#pragma unroll
  for (int e = 0; e < TE; ++e) coef[(size_t)t * TE + e] = 0.0f;
  coef[(size_t)t * TE + i0] = w0;
  coef[(size_t)t * TE + i1] = w1;
}

// ---------------------------------------------------------------------------
// Batched transpose+convert: src [batch][K][N] fp32 -> dst [batch][N][K] bf16
// so B fragments load with the same two-b128 pattern as A fragments.
// ---------------------------------------------------------------------------
__global__ void cvtT_kernel(const float* __restrict__ src, __bf16* __restrict__ dst,
                            int K, int N, long total)
{
  const long per = (long)K * N;
  for (long idx = (long)blockIdx.x * blockDim.x + threadIdx.x; idx < total;
       idx += (long)gridDim.x * blockDim.x) {
    long bb = idx / per;
    long r  = idx - bb * per;
    long n  = r / K;
    long k  = r - n * K;
    dst[idx] = (__bf16)src[bb * per + k * N + n];
  }
}

// ---------------------------------------------------------------------------
// Mean pool over sequence and the small output head (tiny flops, plain VALU).
// ---------------------------------------------------------------------------
__global__ void pool_kernel(const float* __restrict__ x, float* __restrict__ pooled)
{
  int idx = blockIdx.x * blockDim.x + threadIdx.x;
  if (idx >= TB * TD) return;
  int b = idx >> 8, d = idx & 255;
  float s = 0.f;
  for (int t = 0; t < TS; ++t) s += x[((size_t)(b * TS + t)) * TD + d];
  pooled[idx] = s * (1.0f / TS);
}

__global__ void head_kernel(const float* __restrict__ pooled,
                            const float* __restrict__ hw,
                            const float* __restrict__ hb,
                            float* __restrict__ out)
{
  int idx = blockIdx.x * blockDim.x + threadIdx.x;
  if (idx >= TB * TOUT) return;
  int b = idx / TOUT, o = idx - b * TOUT;
  float s = hb[o];
  for (int d = 0; d < TD; ++d) s += pooled[b * TD + d] * hw[d * TOUT + o];
  out[idx] = s;
}

// ---------------------------------------------------------------------------
// Host orchestration
// ---------------------------------------------------------------------------
extern "C" void kernel_launch(void* const* d_in, const int* in_sizes, int n_in,
                              void* d_out, int out_size, void* d_ws, size_t ws_size,
                              hipStream_t stream)
{
  (void)in_sizes; (void)n_in; (void)out_size; (void)ws_size;
  const float* x_in   = (const float*)d_in[0];
  const float* qkv_w  = (const float*)d_in[1];
  const float* qkv_b  = (const float*)d_in[2];
  const float* attn_w = (const float*)d_in[3];
  const float* attn_b = (const float*)d_in[4];
  const float* gate_w = (const float*)d_in[5];
  const float* gate_b = (const float*)d_in[6];
  const float* e_w1   = (const float*)d_in[7];
  const float* e_b1   = (const float*)d_in[8];
  const float* e_w2   = (const float*)d_in[9];
  const float* e_b2   = (const float*)d_in[10];
  const float* ln1_g  = (const float*)d_in[11];
  const float* ln1_b  = (const float*)d_in[12];
  const float* ln2_g  = (const float*)d_in[13];
  const float* ln2_b  = (const float*)d_in[14];
  const float* head_w = (const float*)d_in[15];
  const float* head_b = (const float*)d_in[16];
  float* out = (float*)d_out;

  const int T = TTOK;
  char* ws = (char*)d_ws;
  size_t off = 0;
  auto alloc = [&](size_t bytes) -> void* {
    void* p = ws + off;
    off = (off + bytes + 255) & ~(size_t)255;
    return p;
  };
  float*  xbuf   = (float*)alloc((size_t)T * TD * 4);
  __bf16* xln    = (__bf16*)alloc((size_t)T * TD * 2);
  __bf16* qkv    = (__bf16*)alloc((size_t)T * 3 * TD * 2);
  __bf16* ao     = (__bf16*)alloc((size_t)T * TD * 2);
  __bf16* hbf    = (__bf16*)alloc((size_t)T * TFF * 2);
  float*  coef   = (float*)alloc((size_t)T * TE * 4);
  float*  pooled = (float*)alloc((size_t)TB * TD * 4);
  __bf16* qkvwT  = (__bf16*)alloc((size_t)TL * 3 * TD * TD * 2);
  __bf16* attnwT = (__bf16*)alloc((size_t)TL * TD * TD * 2);
  __bf16* w1T    = (__bf16*)alloc((size_t)TL * TE * TFF * TD * 2);
  __bf16* w2T    = (__bf16*)alloc((size_t)TL * TE * TD * TFF * 2);

  // working residual stream copy (inputs must not be mutated)
  hipMemcpyAsync(xbuf, x_in, (size_t)T * TD * 4, hipMemcpyDeviceToDevice, stream);

  // weights: fp32 [K,N] -> bf16 [N,K]
  cvtT_kernel<<<2048, 256, 0, stream>>>(qkv_w,  qkvwT,  TD,  3 * TD, (long)TL * TD * 3 * TD);
  cvtT_kernel<<<1024, 256, 0, stream>>>(attn_w, attnwT, TD,  TD,     (long)TL * TD * TD);
  cvtT_kernel<<<4096, 256, 0, stream>>>(e_w1,   w1T,    TD,  TFF,    (long)TL * TE * TD * TFF);
  cvtT_kernel<<<4096, 256, 0, stream>>>(e_w2,   w2T,    TFF, TD,     (long)TL * TE * TFF * TD);

  const dim3 gblk(32, 4);
  for (int l = 0; l < TL; ++l) {
    // --- attention block ---
    ln_kernel<<<T, 256, 0, stream>>>(xbuf, ln1_g, ln1_b, xln);
    gemm_wmma<0><<<dim3((3 * TD) / 64, T / 128), gblk, 0, stream>>>(
        xln, TD, qkvwT + (size_t)l * 3 * TD * TD, TD, qkv_b + l * 3 * TD,
        nullptr, qkv, 3 * TD, nullptr, T, 3 * TD, TD);
    attn_kernel<<<dim3(TB * TH, TS / 16), 32, 0, stream>>>(qkv, ao);
    gemm_wmma<1><<<dim3(TD / 64, T / 128), gblk, 0, stream>>>(
        ao, TD, attnwT + (size_t)l * TD * TD, TD, attn_b + l * TD,
        xbuf, nullptr, TD, nullptr, T, TD, TD);

    // --- MoE block (dense all-expert, matching reference; combine fused) ---
    ln_kernel<<<T, 256, 0, stream>>>(xbuf, ln2_g, ln2_b, xln);
    gate_kernel<<<T / 256, 256, 0, stream>>>(
        xln, gate_w + (size_t)l * TD * TE, gate_b + l * TE, coef, T);
    for (int e = 0; e < TE; ++e) {
      const int le = l * TE + e;
      gemm_wmma<2><<<dim3(TFF / 64, T / 128), gblk, 0, stream>>>(
          xln, TD, w1T + (size_t)le * TFF * TD, TD, e_b1 + (size_t)le * TFF,
          nullptr, hbf, TFF, nullptr, T, TFF, TD);
      gemm_wmma<3><<<dim3(TD / 64, T / 128), gblk, 0, stream>>>(
          hbf, TFF, w2T + (size_t)le * TD * TFF, TFF, e_b2 + (size_t)le * TD,
          xbuf, nullptr, TD, coef + e, T, TD, TFF);
    }
  }

  pool_kernel<<<(TB * TD + 255) / 256, 256, 0, stream>>>(xbuf, pooled);
  head_kernel<<<(TB * TOUT + 255) / 256, 256, 0, stream>>>(pooled, head_w, head_b, out);
}
